// QuantizeConv_11819749998901
// MI455X (gfx1250) — 8-bit matrix operand order repaired and checked by reading, compiled, not run on hardware
//
#include <hip/hip_runtime.h>
#include <cstdint>
#include <cstddef>


typedef __attribute__((ext_vector_type(8))) int v8i;

#define NIMG   32
#define CIN    256
#define HDIM   56
#define WDIM   56
#define HWP    (HDIM * WDIM)
#define PDIM   58
#define OCH    256
#define KTOT   2304
#define KTILES 36

#define WS_ALPHA   0
#define WS_MSCALE  1024
#define WS_WPACK   4096
#define WS_XQ8     593920
#define XQ8_BYTES  (NIMG * PDIM * PDIM * 256)

__global__ void k_alpha(const float* __restrict__ x, float* __restrict__ alpha) {
  __shared__ float sm[256];
  const int n = blockIdx.x;
  const float4* xv = (const float4*)(x + (size_t)n * (CIN * HWP));
  const int nv = CIN * HWP / 4;
  float s = 0.f;
  for (int i = threadIdx.x; i < nv; i += 256) {
    float4 v = xv[i];
    s += fabsf(v.x) + fabsf(v.y) + fabsf(v.z) + fabsf(v.w);
  }
  sm[threadIdx.x] = s;
  __syncthreads();
  for (int off = 128; off > 0; off >>= 1) {
    if (threadIdx.x < off) sm[threadIdx.x] += sm[threadIdx.x + off];
    __syncthreads();
  }
  if (threadIdx.x == 0) {
    float a = 2.f * sm[0] / (float)(CIN * HWP);
    alpha[n] = fmaxf(a, 1e-5f);
  }
}

__global__ void k_wscale(const float* __restrict__ w, float* __restrict__ ms) {
  __shared__ float sm[256];
  const int o = blockIdx.x;
  const float* wp = w + (size_t)o * KTOT;
  float s = 0.f;
  for (int i = threadIdx.x; i < KTOT; i += 256) s += fabsf(wp[i]);
  sm[threadIdx.x] = s;
  __syncthreads();
  for (int off = 128; off > 0; off >>= 1) {
    if (threadIdx.x < off) sm[threadIdx.x] += sm[threadIdx.x + off];
    __syncthreads();
  }
  if (threadIdx.x == 0) ms[o] = sm[0] / (float)KTOT;
}

__global__ void k_wpack(const float* __restrict__ w, uint32_t* __restrict__ wp) {
  const int d    = blockIdx.x * 256 + threadIdx.x;
  const int j    = d & 7;
  const int lane = (d >> 3) & 31;
  const int rest = d >> 8;
  const int t    = rest % KTILES;
  const int ob   = rest / KTILES;
  const int m    = lane & 15;
  const int half = lane >> 4;
  const int o    = ob * 16 + m;
  const int kk   = t >> 2;
  const int cb   = t & 3;
  const int kbase = ((j >> 2) << 5) + ((j & 3) << 2) + (half << 4);
  uint32_t packed = 0;
#pragma unroll
  for (int b = 0; b < 4; ++b) {
    const int c = cb * 64 + kbase + b;
    const float v = w[((size_t)o * CIN + c) * 9 + kk];
    const int sg = (v > 0.f) - (v < 0.f);
    packed |= ((uint32_t)(uint8_t)(int8_t)sg) << (8 * b);
  }
  wp[d] = packed;
}

__global__ void k_xzero(uint4* __restrict__ p) {
  const size_t i = (size_t)blockIdx.x * 256 + threadIdx.x;
  const uint4 z = make_uint4(0u, 0u, 0u, 0u);
  p[i] = z;
}

__global__ void k_xquant(const float* __restrict__ x, uint32_t* __restrict__ xq) {
  const size_t idx = (size_t)blockIdx.x * 256 + threadIdx.x;
  const int   c4  = (int)(idx & 63);
  const size_t s  = idx >> 6;
  const int   n   = (int)(s / HWP);
  const int   hw  = (int)(s % HWP);
  const int   h   = hw / WDIM;
  const int   w   = hw - h * WDIM;
  const float* xp = x + ((size_t)n * CIN + c4 * 4) * HWP + hw;
  uint32_t packed = 0;
#pragma unroll
  for (int jj = 0; jj < 4; ++jj) {
    const float v = xp[(size_t)jj * HWP];
    const int sg = (v > 0.f) - (v < 0.f);
    packed |= ((uint32_t)(uint8_t)(int8_t)sg) << (8 * jj);
  }
  xq[((size_t)(n * PDIM + h + 1) * PDIM + (w + 1)) * 64 + c4] = packed;
}

__global__ __launch_bounds__(256)
void k_conv(const uint8_t* __restrict__ xq, const uint8_t* __restrict__ wpk,
            const float* __restrict__ alpha, const float* __restrict__ ms,
            float* __restrict__ out) {
  const int lane = threadIdx.x & 31;
  const int wv   = threadIdx.x >> 5;
  const int wvM  = wv & 1;
  const int wvN  = wv >> 1;
  const int col  = lane & 15;
  const int half = lane >> 4;

  const int ob0   = blockIdx.y * 4 + wvM * 2;
  const int sBase = blockIdx.x * 256 + wvN * 64;

  const uint8_t* pB[4];
#pragma unroll
  for (int jn = 0; jn < 4; ++jn) {
    const int s  = sBase + jn * 16 + col;
    const int n  = s / HWP;
    const int hw = s - n * HWP;
    const int h  = hw / WDIM;
    const int w  = hw - h * WDIM;
    pB[jn] = xq + (((size_t)(n * PDIM + h) * PDIM + w) << 8) + half * 16;
  }

  v8i acc[2][4];
#pragma unroll
  for (int i = 0; i < 2; ++i)
#pragma unroll
    for (int jn = 0; jn < 4; ++jn) acc[i][jn] = (v8i)0;

  const uint8_t* pA0 = wpk + (size_t)(ob0 * KTILES) * 1024 + lane * 32;
  const uint8_t* pA1 = pA0 + (size_t)KTILES * 1024;

#pragma unroll
  for (int kk = 0; kk < 9; ++kk) {
    const int dh   = kk / 3;
    const int dw   = kk - dh * 3;
    const int sOff = (dh * PDIM + dw) << 8;
#pragma unroll
    for (int cb = 0; cb < 4; ++cb) {
      const int t = kk * 4 + cb;
      const v8i a0 = *(const v8i*)(pA0 + t * 1024);
      const v8i a1 = *(const v8i*)(pA1 + t * 1024);

      v8i b[4];
#pragma unroll
      for (int jn = 0; jn < 4; ++jn) {
        const int4 lo = *(const int4*)(pB[jn] + sOff + cb * 64);
        const int4 hi = *(const int4*)(pB[jn] + sOff + cb * 64 + 32);
        v8i bf;
        bf[0] = lo.x; bf[1] = lo.y; bf[2] = lo.z; bf[3] = lo.w;
        bf[4] = hi.x; bf[5] = hi.y; bf[6] = hi.z; bf[7] = hi.w;
        b[jn] = bf;
      }

      acc[0][0] = __builtin_amdgcn_wmma_i32_16x16x64_iu8(true, a0, true, b[0], acc[0][0], false, false);
      acc[0][1] = __builtin_amdgcn_wmma_i32_16x16x64_iu8(true, a0, true, b[1], acc[0][1], false, false);
      acc[0][2] = __builtin_amdgcn_wmma_i32_16x16x64_iu8(true, a0, true, b[2], acc[0][2], false, false);
      acc[0][3] = __builtin_amdgcn_wmma_i32_16x16x64_iu8(true, a0, true, b[3], acc[0][3], false, false);
      acc[1][0] = __builtin_amdgcn_wmma_i32_16x16x64_iu8(true, a1, true, b[0], acc[1][0], false, false);
      acc[1][1] = __builtin_amdgcn_wmma_i32_16x16x64_iu8(true, a1, true, b[1], acc[1][1], false, false);
      acc[1][2] = __builtin_amdgcn_wmma_i32_16x16x64_iu8(true, a1, true, b[2], acc[1][2], false, false);
      acc[1][3] = __builtin_amdgcn_wmma_i32_16x16x64_iu8(true, a1, true, b[3], acc[1][3], false, false);
    }
  }

#pragma unroll
  for (int jn = 0; jn < 4; ++jn) {
    const int s  = sBase + jn * 16 + col;
    const int n  = s / HWP;
    const int hw = s - n * HWP;
    const float a = alpha[n];
#pragma unroll
    for (int i = 0; i < 2; ++i) {
      const int oBase = (ob0 + i) * 16 + half * 8;
#pragma unroll
      for (int r = 0; r < 8; ++r) {
        const int o = oBase + r;
        out[((size_t)n * OCH + o) * HWP + hw] = a * ms[o] * (float)acc[i][jn][r];
      }
    }
  }
}

extern "C" void kernel_launch(void* const* d_in, const int* in_sizes, int n_in,
                              void* d_out, int out_size, void* d_ws, size_t ws_size,
                              hipStream_t stream) {
  const float* x = (const float*)d_in[0];
  const float* w = (const float*)d_in[1];
  float* out = (float*)d_out;

  char* ws = (char*)d_ws;
  float*    alpha = (float*)(ws + WS_ALPHA);
  float*    msc   = (float*)(ws + WS_MSCALE);
  uint32_t* wpk   = (uint32_t*)(ws + WS_WPACK);
  uint32_t* xq    = (uint32_t*)(ws + WS_XQ8);

  k_alpha <<<NIMG,  256, 0, stream>>>(x, alpha);
  k_wscale<<<OCH,   256, 0, stream>>>(w, msc);
  k_wpack <<<576,   256, 0, stream>>>(w, wpk);
  k_xzero <<<XQ8_BYTES / 16 / 256, 256, 0, stream>>>((uint4*)xq);
  k_xquant<<<25088, 256, 0, stream>>>(x, xq);

  dim3 grid(392  , 4  );
  k_conv<<<grid, 256, 0, stream>>>((const uint8_t*)xq, (const uint8_t*)wpk,
                                   alpha, msc, out);
}
